// DiffnetPPModel_26895085208196
// MI455X (gfx1250) — matrix operand order repaired and checked by reading, not compiled, not run on hardware
//
#include <hip/hip_runtime.h>
#include <hip/hip_bf16.h>

typedef __attribute__((ext_vector_type(16))) _Float16 v16h;
typedef __attribute__((ext_vector_type(8)))  _Float16 v8h;
typedef __attribute__((ext_vector_type(8)))  float    v8f;

#define GAT_SLOPE 0.2f
#define MLP_SLOPE 0.01f
#define NEG_INF  (-3.402823466e38f)

__device__ __forceinline__ float leakyf(float x, float s) { return x > 0.f ? x : s * x; }

__device__ __forceinline__ void atomicMaxF(float* addr, float v) {
    int* ia = (int*)addr;
    int old = *ia;
    while (__int_as_float(old) < v) {
        int assumed = old;
        old = atomicCAS(ia, assumed, __float_as_int(v));
        if (old == assumed) break;
    }
}

// ---------------------------------------------------------------------------
// WMMA GEMM: Y[N,64] = X[N,K] @ W[K,64] + bias,  K in {64,128}.
// For K=128 the input columns are split: cols [0,64) from X1 (stride ldx1),
// cols [64,128) from X2 (stride ldx2)  -> implements concat([X1,X2]) @ W.
// Block = 128 threads = 4 waves; each wave computes a 16x64 output strip with
// 4 accumulators of v_wmma_f32_16x16x32_f16 per 32-wide k-step.
// Weight tile is staged in LDS *transposed* (Wl[n][k], f16, padded stride)
// so each B fragment is two contiguous 16B ds_load_b128 per lane.
// ---------------------------------------------------------------------------
#define KP 136   // padded K stride in halves (272B = 68 dwords; bank = 4*lane mod 64)

__global__ __launch_bounds__(128)
void k_gemm_wmma(const float* __restrict__ X1, int ldx1,
                 const float* __restrict__ X2, int ldx2,
                 const float* __restrict__ W,      // [K,64] row-major
                 const float* __restrict__ bias,   // [64]
                 int K, int N,
                 float* __restrict__ Y, int ldy)
{
    __shared__ __align__(16) _Float16 Wl[64 * KP];   // transposed: Wl[n*KP + k]
    const int tid = threadIdx.x;
    const int kw = K * 64;
    for (int i = tid; i < kw; i += 128) {
        const int k = i >> 6, n = i & 63;
        Wl[n * KP + k] = (_Float16)W[i];
    }
    __syncthreads();

    const int lane = tid & 31;          // wave32
    const int wave = tid >> 5;
    const int l16  = lane & 15;
    const int half = lane >> 4;         // 0 or 1
    const int rowBase = blockIdx.x * 64 + wave * 16;
    if (rowBase >= N) return;

    v8f acc[4];
#pragma unroll
    for (int n = 0; n < 4; ++n)
#pragma unroll
        for (int j = 0; j < 8; ++j) acc[n][j] = 0.f;

    int rowA = rowBase + l16;
    if (rowA >= N) rowA = N - 1;        // clamp; garbage rows are not stored

    const int ksteps = K >> 5;
    for (int ks = 0; ks < ksteps; ++ks) {
        const int kb = ks * 32;
        const float* Xp; int ldx; int kb2;
        if (kb < 64) { Xp = X1; ldx = ldx1; kb2 = kb; }
        else         { Xp = X2; ldx = ldx2; kb2 = kb - 64; }
        // A fragment (16x32 f16): lanes 0-15 rows M=0..15 K={0..7,16..23},
        // lanes 16-31 same rows K={8..15,24..31}
        const float* xr = Xp + (size_t)rowA * ldx + kb2 + half * 8;
        float4 f0 = *(const float4*)(xr + 0);
        float4 f1 = *(const float4*)(xr + 4);
        float4 f2 = *(const float4*)(xr + 16);
        float4 f3 = *(const float4*)(xr + 20);
        v16h a;
        a[0]=(_Float16)f0.x;  a[1]=(_Float16)f0.y;  a[2]=(_Float16)f0.z;  a[3]=(_Float16)f0.w;
        a[4]=(_Float16)f1.x;  a[5]=(_Float16)f1.y;  a[6]=(_Float16)f1.z;  a[7]=(_Float16)f1.w;
        a[8]=(_Float16)f2.x;  a[9]=(_Float16)f2.y;  a[10]=(_Float16)f2.z; a[11]=(_Float16)f2.w;
        a[12]=(_Float16)f3.x; a[13]=(_Float16)f3.y; a[14]=(_Float16)f3.z; a[15]=(_Float16)f3.w;

#pragma unroll
        for (int n = 0; n < 4; ++n) {
            // B fragment (32x16 f16): lanes 0-15 N=lane K=kb+{0..7,16..23},
            // lanes 16-31 N=lane-16 K=kb+{8..15,24..31} -> two 8-runs in transposed Wl
            const _Float16* wp = &Wl[(size_t)(n * 16 + l16) * KP + kb + half * 8];
            const v8h b0 = *(const v8h*)(wp + 0);
            const v8h b1 = *(const v8h*)(wp + 16);
            v16h b;
#pragma unroll
            for (int j = 0; j < 8; ++j) { b[j] = b0[j]; b[j + 8] = b1[j]; }
            acc[n] = __builtin_amdgcn_wmma_f32_16x16x32_f16(
                false, a, false, b, (short)0, acc[n], false, false);
        }
    }

    // C/D layout: VGPR r -> M=r (lanes 0-15) / M=r+8 (lanes 16-31), N=l16
#pragma unroll
    for (int n = 0; n < 4; ++n) {
        const int col = n * 16 + l16;
        const float bv = bias[col];
#pragma unroll
        for (int r = 0; r < 8; ++r) {
            const int row = rowBase + r + half * 8;
            if (row < N) Y[(size_t)row * ldy + col] = acc[n][r] + bv;
        }
    }
}

// ---------------------------------------------------------------------------
__global__ void k_fill(float* __restrict__ p, float v, long n) {
    long i = (long)blockIdx.x * blockDim.x + threadIdx.x;
    if (i < n) p[i] = v;
}

__global__ void k_copy64(const float* __restrict__ src, int lds,
                         float* __restrict__ dst, int ldd, int N) {
    long t = (long)blockIdx.x * blockDim.x + threadIdx.x;
    if (t >= (long)N * 64) return;
    int row = (int)(t >> 6), col = (int)(t & 63);
    dst[(size_t)row * ldd + col] = src[(size_t)row * lds + col];
}

// e = a . leaky_relu(fs[src] + fd[dst]); running segment max into m[dst]
__global__ void k_edge_score(const float* __restrict__ fs, const float* __restrict__ fd,
                             const float* __restrict__ avec,
                             const int* __restrict__ src, const int* __restrict__ dst,
                             float* __restrict__ e, float* __restrict__ m, int E) {
    long t = (long)blockIdx.x * blockDim.x + threadIdx.x;
    int edge = (int)(t >> 4), l = (int)(t & 15);
    if (edge >= E) return;
    int s = src[edge], d = dst[edge];
    float4 a4 = *(const float4*)(avec + l * 4);
    float4 u4 = *(const float4*)(fs + (size_t)s * 64 + l * 4);
    float4 v4 = *(const float4*)(fd + (size_t)d * 64 + l * 4);
    float sum = a4.x * leakyf(u4.x + v4.x, GAT_SLOPE)
              + a4.y * leakyf(u4.y + v4.y, GAT_SLOPE)
              + a4.z * leakyf(u4.z + v4.z, GAT_SLOPE)
              + a4.w * leakyf(u4.w + v4.w, GAT_SLOPE);
    for (int off = 8; off > 0; off >>= 1) sum += __shfl_xor(sum, off, 16);
    if (l == 0) { e[edge] = sum; atomicMaxF(&m[d], sum); }
}

__global__ void k_edge_exp(const float* __restrict__ e, float* __restrict__ ex,
                           const float* __restrict__ m, float* __restrict__ den,
                           const int* __restrict__ dst, int E) {
    int i = blockIdx.x * blockDim.x + threadIdx.x;
    if (i >= E) return;
    int d = dst[i];
    float v = __expf(e[i] - m[d]);
    ex[i] = v;
    atomicAdd(&den[d], v);
}

// out[dst] += (ex/den[dst]) * fs[src]
__global__ void k_edge_scatter(const float* __restrict__ fs,
                               const int* __restrict__ src, const int* __restrict__ dst,
                               const float* __restrict__ ex, const float* __restrict__ den,
                               float* __restrict__ out, int ldo, int E) {
    long t = (long)blockIdx.x * blockDim.x + threadIdx.x;
    int edge = (int)(t >> 4), l = (int)(t & 15);
    if (edge >= E) return;
    int s = src[edge], d = dst[edge];
    float alpha = ex[edge] / den[d];
    float4 u4 = *(const float4*)(fs + (size_t)s * 64 + l * 4);
    float* op = out + (size_t)d * ldo + l * 4;
    atomicAdd(op + 0, alpha * u4.x);
    atomicAdd(op + 1, alpha * u4.y);
    atomicAdd(op + 2, alpha * u4.z);
    atomicAdd(op + 3, alpha * u4.w);
}

// gamma-softmax gating: u_out = g0*p + g1*q + u_in
__global__ void k_gate(const float* __restrict__ hA, const float* __restrict__ hB,
                       const float* __restrict__ W2a, const float* __restrict__ b2a,
                       const float* __restrict__ W2b, const float* __restrict__ b2b,
                       const float* __restrict__ p, const float* __restrict__ q,
                       const float* __restrict__ uin, int ldu,
                       float* __restrict__ uout, int N) {
    long t = (long)blockIdx.x * blockDim.x + threadIdx.x;
    int row = (int)(t >> 4), l = (int)(t & 15);
    if (row >= N) return;
    float4 a4 = *(const float4*)(hA + (size_t)row * 64 + l * 4);
    float4 wa = *(const float4*)(W2a + l * 4);
    float4 b4 = *(const float4*)(hB + (size_t)row * 64 + l * 4);
    float4 wb = *(const float4*)(W2b + l * 4);
    float sA = a4.x * wa.x + a4.y * wa.y + a4.z * wa.z + a4.w * wa.w;
    float sB = b4.x * wb.x + b4.y * wb.y + b4.z * wb.z + b4.w * wb.w;
    for (int off = 8; off > 0; off >>= 1) {
        sA += __shfl_xor(sA, off, 16);
        sB += __shfl_xor(sB, off, 16);
    }
    float s_inf = leakyf(sA + b2a[0], MLP_SLOPE);
    float s_int = leakyf(sB + b2b[0], MLP_SLOPE);
    float mx = fmaxf(s_inf, s_int);
    float e0 = __expf(s_inf - mx), e1 = __expf(s_int - mx);
    float g0 = e0 / (e0 + e1), g1 = 1.f - g0;
    float4 pv = *(const float4*)(p + (size_t)row * 64 + l * 4);
    float4 qv = *(const float4*)(q + (size_t)row * 64 + l * 4);
    float4 uv = *(const float4*)(uin + (size_t)row * ldu + l * 4);
    float4 r;
    r.x = g0 * pv.x + g1 * qv.x + uv.x;
    r.y = g0 * pv.y + g1 * qv.y + uv.y;
    r.z = g0 * pv.z + g1 * qv.z + uv.z;
    r.w = g0 * pv.w + g1 * qv.w + uv.w;
    *(float4*)(uout + (size_t)row * ldu + l * 4) = r;
}

// out[e] = dot(hu[src[e]], hi[dst[e]]) over RD dims (RD multiple of 64)
__global__ void k_pred(const float* __restrict__ hu, const float* __restrict__ hi,
                       const int* __restrict__ src, const int* __restrict__ dst,
                       float* __restrict__ out, int RD, int E) {
    long t = (long)blockIdx.x * blockDim.x + threadIdx.x;
    int edge = (int)(t >> 4), l = (int)(t & 15);
    if (edge >= E) return;
    const int chunk = RD >> 4;                 // floats per lane (multiple of 4)
    const float* ur = hu + (size_t)src[edge] * RD + l * chunk;
    const float* ir = hi + (size_t)dst[edge] * RD + l * chunk;
    float s = 0.f;
    for (int j = 0; j < chunk; j += 4) {
        float4 a = *(const float4*)(ur + j);
        float4 b = *(const float4*)(ir + j);
        s += a.x * b.x + a.y * b.y + a.z * b.z + a.w * b.w;
    }
    for (int off = 8; off > 0; off >>= 1) s += __shfl_xor(s, off, 16);
    if (l == 0) out[edge] = s;
}

// ---------------------------------------------------------------------------
extern "C" void kernel_launch(void* const* d_in, const int* in_sizes, int n_in,
                              void* d_out, int out_size, void* d_ws, size_t ws_size,
                              hipStream_t stream) {
    (void)n_in; (void)out_size; (void)ws_size;
    const int D = 64;
    const float* user_emb = (const float*)d_in[0];
    const float* item_emb = (const float*)d_in[1];
    const int* rate_src  = (const int*)d_in[2];
    const int* rate_dst  = (const int*)d_in[3];
    const int* trust_src = (const int*)d_in[4];
    const int* trust_dst = (const int*)d_in[5];
    const int* pos_src   = (const int*)d_in[6];
    const int* pos_dst   = (const int*)d_in[7];
    const int* neg_src   = (const int*)d_in[8];
    const int* neg_dst   = (const int*)d_in[9];
    const float* gW[3][5];   // [rate,rb,tr] x [Ws,bs,Wd,bd,a]
    for (int g = 0; g < 3; ++g)
        for (int k = 0; k < 5; ++k)
            gW[g][k] = (const float*)d_in[10 + g * 5 + k];
    const float* inf_W1 = (const float*)d_in[25];
    const float* inf_b1 = (const float*)d_in[26];
    const float* inf_W2 = (const float*)d_in[27];
    const float* inf_b2 = (const float*)d_in[28];
    const float* int_W1 = (const float*)d_in[29];
    const float* int_b1 = (const float*)d_in[30];
    const float* int_W2 = (const float*)d_in[31];
    const float* int_b2 = (const float*)d_in[32];

    const int U    = in_sizes[0] / D;
    const int I    = in_sizes[1] / D;
    const int E_ui = in_sizes[2];
    const int E_uu = in_sizes[4];
    const int E_pr = in_sizes[6];
    const int L    = in_sizes[10] / (D * D);
    const int RD   = (L + 1) * D;
    const int Nmax = (U > I) ? U : I;
    const int Emax = (E_ui > E_uu) ? E_ui : E_uu;

    float* w = (float*)d_ws;
    float* res_u = w;  w += (size_t)U * RD;
    float* res_i = w;  w += (size_t)I * RD;
    float* pbuf  = w;  w += (size_t)U * D;
    float* qbuf  = w;  w += (size_t)U * D;
    float* tA    = w;  w += (size_t)Nmax * D;
    float* tB    = w;  w += (size_t)Nmax * D;
    float* ebuf  = w;  w += (size_t)Emax;
    float* exbuf = w;  w += (size_t)Emax;
    float* mbuf  = w;  w += (size_t)Nmax;
    float* denbuf= w;  w += (size_t)Nmax;

    const int TB = 256;
    auto cdiv = [](long a, long b) { return (int)((a + b - 1) / b); };

    // layer 0 residual slices
    k_copy64<<<cdiv((long)U * 64, TB), TB, 0, stream>>>(user_emb, D, res_u, RD, U);
    k_copy64<<<cdiv((long)I * 64, TB), TB, 0, stream>>>(item_emb, D, res_i, RD, I);

    for (int l = 0; l < L; ++l) {
        const float* uS = res_u + (size_t)l * D;
        const float* iS = res_i + (size_t)l * D;
        float* uO = res_u + (size_t)(l + 1) * D;
        float* iO = res_i + (size_t)(l + 1) * D;
        const size_t oW = (size_t)l * D * D, oB = (size_t)l * D;

        // ---- rate GATv2: user -> item (+residual via init-copy) ----
        k_gemm_wmma<<<cdiv(U,64),128,0,stream>>>(uS, RD, nullptr, 0, gW[0][0]+oW, gW[0][1]+oB, D, U, tA, D);
        k_gemm_wmma<<<cdiv(I,64),128,0,stream>>>(iS, RD, nullptr, 0, gW[0][2]+oW, gW[0][3]+oB, D, I, tB, D);
        k_copy64<<<cdiv((long)I*64,TB),TB,0,stream>>>(iS, RD, iO, RD, I);
        k_fill<<<cdiv(I,TB),TB,0,stream>>>(mbuf, NEG_INF, I);
        k_fill<<<cdiv(I,TB),TB,0,stream>>>(denbuf, 0.f, I);
        k_edge_score<<<cdiv((long)E_ui*16,TB),TB,0,stream>>>(tA, tB, gW[0][4]+oB, rate_src, rate_dst, ebuf, mbuf, E_ui);
        k_edge_exp<<<cdiv(E_ui,TB),TB,0,stream>>>(ebuf, exbuf, mbuf, denbuf, rate_dst, E_ui);
        k_edge_scatter<<<cdiv((long)E_ui*16,TB),TB,0,stream>>>(tA, rate_src, rate_dst, exbuf, denbuf, iO, RD, E_ui);

        // ---- rb GATv2: item -> user (reversed rate edges) -> q ----
        k_gemm_wmma<<<cdiv(I,64),128,0,stream>>>(iS, RD, nullptr, 0, gW[1][0]+oW, gW[1][1]+oB, D, I, tA, D);
        k_gemm_wmma<<<cdiv(U,64),128,0,stream>>>(uS, RD, nullptr, 0, gW[1][2]+oW, gW[1][3]+oB, D, U, tB, D);
        k_fill<<<cdiv((long)U*D,TB),TB,0,stream>>>(qbuf, 0.f, (long)U * D);
        k_fill<<<cdiv(U,TB),TB,0,stream>>>(mbuf, NEG_INF, U);
        k_fill<<<cdiv(U,TB),TB,0,stream>>>(denbuf, 0.f, U);
        k_edge_score<<<cdiv((long)E_ui*16,TB),TB,0,stream>>>(tA, tB, gW[1][4]+oB, rate_dst, rate_src, ebuf, mbuf, E_ui);
        k_edge_exp<<<cdiv(E_ui,TB),TB,0,stream>>>(ebuf, exbuf, mbuf, denbuf, rate_src, E_ui);
        k_edge_scatter<<<cdiv((long)E_ui*16,TB),TB,0,stream>>>(tA, rate_dst, rate_src, exbuf, denbuf, qbuf, D, E_ui);

        // ---- trust GATv2: user -> user -> p ----
        k_gemm_wmma<<<cdiv(U,64),128,0,stream>>>(uS, RD, nullptr, 0, gW[2][0]+oW, gW[2][1]+oB, D, U, tA, D);
        k_gemm_wmma<<<cdiv(U,64),128,0,stream>>>(uS, RD, nullptr, 0, gW[2][2]+oW, gW[2][3]+oB, D, U, tB, D);
        k_fill<<<cdiv((long)U*D,TB),TB,0,stream>>>(pbuf, 0.f, (long)U * D);
        k_fill<<<cdiv(U,TB),TB,0,stream>>>(mbuf, NEG_INF, U);
        k_fill<<<cdiv(U,TB),TB,0,stream>>>(denbuf, 0.f, U);
        k_edge_score<<<cdiv((long)E_uu*16,TB),TB,0,stream>>>(tA, tB, gW[2][4]+oB, trust_src, trust_dst, ebuf, mbuf, E_uu);
        k_edge_exp<<<cdiv(E_uu,TB),TB,0,stream>>>(ebuf, exbuf, mbuf, denbuf, trust_dst, E_uu);
        k_edge_scatter<<<cdiv((long)E_uu*16,TB),TB,0,stream>>>(tA, trust_src, trust_dst, exbuf, denbuf, pbuf, D, E_uu);

        // ---- gating MLPs (K=128 concat GEMMs on WMMA) + gamma-softmax ----
        k_gemm_wmma<<<cdiv(U,64),128,0,stream>>>(uS, RD, pbuf, D, inf_W1 + (size_t)l*2*D*D, inf_b1 + oB, 2*D, U, tA, D);
        k_gemm_wmma<<<cdiv(U,64),128,0,stream>>>(uS, RD, qbuf, D, int_W1 + (size_t)l*2*D*D, int_b1 + oB, 2*D, U, tB, D);
        k_gate<<<cdiv((long)U*16,TB),TB,0,stream>>>(tA, tB,
            inf_W2 + oB, inf_b2 + l, int_W2 + oB, int_b2 + l,
            pbuf, qbuf, uS, RD, uO, U);
    }

    // ---- final predictions: pos then neg, concatenated flat ----
    float* out = (float*)d_out;
    k_pred<<<cdiv((long)E_pr*16,TB),TB,0,stream>>>(res_u, res_i, pos_src, pos_dst, out, RD, E_pr);
    k_pred<<<cdiv((long)E_pr*16,TB),TB,0,stream>>>(res_u, res_i, neg_src, neg_dst, out + E_pr, RD, E_pr);
}
